// SparseSelfAttention_69827578298918
// MI455X (gfx1250) — matrix operand order repaired and checked by reading, not compiled, not run on hardware
//
#include <hip/hip_runtime.h>
#include <hip/hip_bf16.h>

// ---------------------------------------------------------------------------
// Sparse self-attention for MI455X (gfx1250, wave32, WMMA bf16 16x16x32,
// async global->LDS copies with double buffering)
//   B=2, T=8192, D=768, BS=64 (window 3*BS=192), G=64 global queries
// ---------------------------------------------------------------------------

#define TB 8192
#define DD_ 768
#define NB 128          // T / BS
#define MTOT 16384      // B * T

typedef __attribute__((ext_vector_type(16))) __bf16 v16bf;
typedef __attribute__((ext_vector_type(8)))  float  v8f;

__device__ __forceinline__ unsigned short f2bf(float f) {
    unsigned int u = __float_as_uint(f);
    u += 0x7FFFu + ((u >> 16) & 1u);       // round-to-nearest-even
    return (unsigned short)(u >> 16);
}

__device__ __forceinline__ v8f wmma_bf16(v16bf a, v16bf b, v8f c) {
    return __builtin_amdgcn_wmma_f32_16x16x32_bf16(
        false, a, false, b, (short)0, c, false, false);
}

// Async DMA: 16 bytes per lane, global -> LDS, tracked by ASYNCcnt.
__device__ __forceinline__ void async_b128(void* lds, const void* g) {
    asm volatile("global_load_async_to_lds_b128 %0, %1, off"
                 :: "v"((unsigned int)(unsigned long long)lds),
                    "v"((unsigned long long)g)
                 : "memory");
}
#define WAIT_ASYNC(N) asm volatile("s_wait_asynccnt " #N ::: "memory")

// A fragment (16x32 bf16, MxK): lane = M row; per ISA table K interleave.
__device__ __forceinline__ v16bf load_frag_a(const unsigned short* s, int rowBase,
                                             int ld, int lane) {
    int h = (lane >> 4) & 1, lr = lane & 15;
    const unsigned short* p = s + (size_t)(rowBase + lr) * ld;
    union { v16bf v; unsigned int u[8]; } f;
#pragma unroll
    for (int j = 0; j < 8; ++j) {
        int kb = ((j & 4) ? 16 : 0) + (h << 3) + ((j & 3) << 1);
        f.u[j] = *(const unsigned int*)(p + kb);
    }
    return f.v;
}

// B fragment (32x16 bf16, KxN): lane = N col; per ISA table K interleave (same as A).
__device__ __forceinline__ v16bf load_frag_b(const unsigned short* s, int rowBase,
                                             int ld, int lane) {
    int h = (lane >> 4) & 1, lr = lane & 15;
    const unsigned short* p = s + (size_t)(rowBase + lr) * ld;
    union { v16bf v; unsigned int u[8]; } f;
#pragma unroll
    for (int j = 0; j < 8; ++j) {
        int kb = ((j & 4) ? 16 : 0) + (h << 3) + ((j & 3) << 1);
        f.u[j] = *(const unsigned int*)(p + kb);
    }
    return f.v;
}

// ---------------------------------------------------------------------------
// f32 -> bf16 conversion (x4 vectorized)
// ---------------------------------------------------------------------------
__global__ void cvt_f32_bf16(const float* __restrict__ in,
                             unsigned short* __restrict__ out, int n4) {
    int i = blockIdx.x * blockDim.x + threadIdx.x;
    if (i < n4) {
        float4 f = ((const float4*)in)[i];
        unsigned long long pk =
            (unsigned long long)f2bf(f.x)         |
            ((unsigned long long)f2bf(f.y) << 16) |
            ((unsigned long long)f2bf(f.z) << 32) |
            ((unsigned long long)f2bf(f.w) << 48);
        ((unsigned long long*)out)[i] = pk;
    }
}

// ---------------------------------------------------------------------------
// C[M,N] = A[M,K] @ W[N,K]^T + bias  (bf16 in, f32 accum, bf16/f32 out)
// 64x64 tile / WG, BK=32, double-buffered async staging.
// ---------------------------------------------------------------------------
__global__ __launch_bounds__(256) void gemm_bias(
    const unsigned short* __restrict__ A,
    const unsigned short* __restrict__ W,
    const float* __restrict__ bias,
    void* __restrict__ Cout, int M, int N, int K, int out_bf16) {
    __shared__ unsigned short As[2][64 * 32];
    __shared__ unsigned short Ws[2][64 * 32];
    int tid = threadIdx.x, lane = tid & 31, w = tid >> 5;
    int m0 = blockIdx.x * 64, n0 = blockIdx.y * 64;
    int mrow = (w & 3) * 16, ncol = (w >> 2) * 32;
    int row4 = tid >> 2, ko4 = (tid & 3) * 8;
    const unsigned short* gA = A + (size_t)(m0 + row4) * K + ko4;
    const unsigned short* gW = W + (size_t)(n0 + row4) * K + ko4;
    unsigned short* lA[2] = {&As[0][row4 * 32 + ko4], &As[1][row4 * 32 + ko4]};
    unsigned short* lW[2] = {&Ws[0][row4 * 32 + ko4], &Ws[1][row4 * 32 + ko4]};

    v8f c0 = {0.f, 0.f, 0.f, 0.f, 0.f, 0.f, 0.f, 0.f};
    v8f c1 = c0;

    int nk = K >> 5;
    async_b128(lA[0], gA);
    async_b128(lW[0], gW);
    for (int t = 0; t < nk; ++t) {
        int cur = t & 1;
        if (t + 1 < nk) {
            async_b128(lA[cur ^ 1], gA + (t + 1) * 32);
            async_b128(lW[cur ^ 1], gW + (t + 1) * 32);
            WAIT_ASYNC(2);
        } else {
            WAIT_ASYNC(0);
        }
        __syncthreads();
        v16bf a  = load_frag_a(As[cur], mrow, 32, lane);
        v16bf b0 = load_frag_b(Ws[cur], ncol, 32, lane);
        v16bf b1 = load_frag_b(Ws[cur], ncol + 16, 32, lane);
        c0 = wmma_bf16(a, b0, c0);
        c1 = wmma_bf16(a, b1, c1);
        __syncthreads();
    }

    int h = (lane >> 4) & 1, lr = lane & 15;
    float bias0 = bias[n0 + ncol + lr];
    float bias1 = bias[n0 + ncol + 16 + lr];
#pragma unroll
    for (int r = 0; r < 8; ++r) {
        int m = m0 + mrow + r + 8 * h;
        int n = n0 + ncol + lr;
        float v0 = c0[r] + bias0;
        float v1 = c1[r] + bias1;
        if (out_bf16) {
            unsigned short* O = (unsigned short*)Cout;
            O[(size_t)m * N + n]      = f2bf(v0);
            O[(size_t)m * N + n + 16] = f2bf(v1);
        } else {
            float* O = (float*)Cout;
            O[(size_t)m * N + n]      = v0;
            O[(size_t)m * N + n + 16] = v1;
        }
    }
}

// ---------------------------------------------------------------------------
// Local windowed attention: one WG per (batch, 64-row query block)
// dyn smem: Qs[2] 8KB | Ks[2] 24KB | Ps 24KB | Vt 24KB | S 48KB | dump = 131200B
// ---------------------------------------------------------------------------
__global__ __launch_bounds__(256) void local_attn(
    const unsigned short* __restrict__ Q,
    const unsigned short* __restrict__ Kk,
    const unsigned short* __restrict__ V,
    unsigned short* __restrict__ attn) {
    extern __shared__ unsigned char smem_raw[];
    unsigned short* Qs0 = (unsigned short*)smem_raw;        // 64*32
    unsigned short* Qs1 = Qs0 + 2048;
    unsigned short* Ks0 = Qs1 + 2048;                       // 192*32
    unsigned short* Ks1 = Ks0 + 6144;
    unsigned short* Ps  = Ks1 + 6144;                       // 64*192
    unsigned short* Vt  = Ps + 12288;                       // Vt[d][k], ld=192
    float* S = (float*)(Vt + 12288);                        // 64*192 f32
    unsigned short* dump = (unsigned short*)(S + 12288);    // 64 (scratch sink)
    unsigned short* QsB[2] = {Qs0, Qs1};
    unsigned short* KsB[2] = {Ks0, Ks1};

    int tid = threadIdx.x, lane = tid & 31, w = tid >> 5;
    int bb = blockIdx.x / NB, nB = blockIdx.x % NB;
    const float scale = 0.03608439182435161f;               // 1/sqrt(768)
    size_t qbase = ((size_t)bb * TB + (size_t)nB * 64) * DD_;
    int h = (lane >> 4) & 1, lr = lane & 15;
    int row4 = tid >> 2, ko4 = (tid & 3) * 8;

    // precompute K-window staging chunks (3 per thread; 192 rows x 4 b128)
    const unsigned short* gQ = Q + qbase + (size_t)row4 * DD_ + ko4;
    int kj[3], kko[3]; bool kval[3]; const unsigned short* gK[3];
#pragma unroll
    for (int c = 0; c < 3; ++c) {
        int i = tid + c * 256;
        int j = i >> 2, ko2 = (i & 3) * 8;
        int pos = (nB - 1) * 64 + j;
        bool v = (pos >= 0 && pos < TB);
        kj[c] = j; kko[c] = ko2; kval[c] = v;
        gK[c] = Kk + ((size_t)bb * TB + (v ? pos : 0)) * DD_ + ko2;
    }
    // pre-zero invalid window rows in both buffers (only edge blocks have any)
#pragma unroll
    for (int c = 0; c < 3; ++c) {
        if (!kval[c]) {
            uint4 z = {0u, 0u, 0u, 0u};
            *(uint4*)(Ks0 + kj[c] * 32 + kko[c]) = z;
            *(uint4*)(Ks1 + kj[c] * 32 + kko[c]) = z;
        }
    }

    // ---- Phase 1: S = Q @ Kwin^T (double-buffered async pipeline) -----
    int mrow = (w & 3) * 16, nbase = (w >> 2) * 96;
    v8f acc[6];
#pragma unroll
    for (int t = 0; t < 6; ++t) acc[t] = (v8f){0.f,0.f,0.f,0.f,0.f,0.f,0.f,0.f};

    auto issue_tile = [&](int buf, int kt) {
        async_b128(QsB[buf] + row4 * 32 + ko4, gQ + kt);
#pragma unroll
        for (int c = 0; c < 3; ++c) {
            unsigned short* dst = kval[c] ? (KsB[buf] + kj[c] * 32 + kko[c]) : dump;
            async_b128(dst, gK[c] + kt);
        }
    };

    issue_tile(0, 0);
    for (int t = 0; t < 24; ++t) {
        int cur = t & 1;
        if (t < 23) { issue_tile(cur ^ 1, (t + 1) * 32); WAIT_ASYNC(4); }
        else        { WAIT_ASYNC(0); }
        __syncthreads();
        v16bf a = load_frag_a(QsB[cur], mrow, 32, lane);
#pragma unroll
        for (int s = 0; s < 6; ++s) {
            v16bf b = load_frag_b(KsB[cur], nbase + s * 16, 32, lane);
            acc[s] = wmma_bf16(a, b, acc[s]);
        }
        __syncthreads();
    }
#pragma unroll
    for (int s = 0; s < 6; ++s)
#pragma unroll
        for (int r = 0; r < 8; ++r)
            S[(mrow + r + 8 * h) * 192 + nbase + s * 16 + lr] = acc[s][r] * scale;
    __syncthreads();

    // ---- softmax over each row (mask invalid window positions) --------
    if (tid < 64) {
        int row = tid;
        float mx = -__builtin_inff();
        for (int j = 0; j < 192; ++j) {
            int pos = (nB - 1) * 64 + j;
            float s = (pos >= 0 && pos < TB) ? S[row * 192 + j] : -__builtin_inff();
            S[row * 192 + j] = s;
            mx = fmaxf(mx, s);
        }
        float sum = 0.f;
        for (int j = 0; j < 192; ++j) {
            float e = __expf(S[row * 192 + j] - mx);
            S[row * 192 + j] = e;
            sum += e;
        }
        float inv = 1.0f / sum;
        for (int j = 0; j < 192; ++j)
            Ps[row * 192 + j] = f2bf(S[row * 192 + j] * inv);
    }
    __syncthreads();

    // ---- Phase 2: O = P @ Vwin, per 64-wide d tile --------------------
    int mrow2 = (w & 3) * 16, ncol2 = (w >> 2) * 32;
    for (int dt = 0; dt < 12; ++dt) {
        // stage V window transposed: Vt[d][k]  (192 rows x 8 b128 chunks)
        for (int i = tid; i < 1536; i += 256) {
            int j = i >> 3, dc = (i & 7) * 8;
            int pos = (nB - 1) * 64 + j;
            unsigned short tmp[8];
            if (pos >= 0 && pos < TB) {
                *(uint4*)tmp = *(const uint4*)(V + ((size_t)bb * TB + pos) * DD_ +
                                               dt * 64 + dc);
            } else {
#pragma unroll
                for (int e = 0; e < 8; ++e) tmp[e] = 0;
            }
#pragma unroll
            for (int e = 0; e < 8; ++e) Vt[(dc + e) * 192 + j] = tmp[e];
        }
        __syncthreads();
        v8f o0 = {0.f,0.f,0.f,0.f,0.f,0.f,0.f,0.f};
        v8f o1 = o0;
#pragma unroll
        for (int kk = 0; kk < 192; kk += 32) {
            v16bf a  = load_frag_a(Ps + kk, mrow2, 192, lane);
            v16bf b0 = load_frag_b(Vt + kk, ncol2, 192, lane);
            v16bf b1 = load_frag_b(Vt + kk, ncol2 + 16, 192, lane);
            o0 = wmma_bf16(a, b0, o0);
            o1 = wmma_bf16(a, b1, o1);
        }
#pragma unroll
        for (int r = 0; r < 8; ++r) {
            int trow = nB * 64 + mrow2 + r + 8 * h;
            size_t base = ((size_t)bb * TB + trow) * DD_ + dt * 64 + ncol2 + lr;
            attn[base]      = f2bf(o0[r]);
            attn[base + 16] = f2bf(o1[r]);
        }
        __syncthreads();
    }
}

// ---------------------------------------------------------------------------
// Global attention, first G=64 queries: one WG per (batch, 64-wide d tile)
// dyn smem: Qs 96KB | Ks[2] 8KB | Ps 8KB | Vt 8KB | S 16KB | stats = 140032B
// ---------------------------------------------------------------------------
__global__ __launch_bounds__(256) void global_attn(
    const unsigned short* __restrict__ Q,
    const unsigned short* __restrict__ Kk,
    const unsigned short* __restrict__ V,
    unsigned short* __restrict__ attn) {
    extern __shared__ unsigned char smem_raw[];
    unsigned short* Qs  = (unsigned short*)smem_raw;   // 64*768
    unsigned short* Ks0 = Qs + 49152;                  // 64*32
    unsigned short* Ks1 = Ks0 + 2048;
    unsigned short* Ps  = Ks1 + 2048;                  // 64*64
    unsigned short* Vt  = Ps + 4096;                   // Vt[d][k], ld=64
    float* S        = (float*)(Vt + 4096);             // 64*64
    float* rowmax   = S + 4096;
    float* rowsum   = rowmax + 64;
    float* rowscale = rowsum + 64;
    unsigned short* KsB[2] = {Ks0, Ks1};

    int tid = threadIdx.x, lane = tid & 31, w = tid >> 5;
    int bb = blockIdx.x / 12, dt = blockIdx.x % 12;
    const float scale = 0.03608439182435161f;
    int h = (lane >> 4) & 1, lr = lane & 15;
    int row4 = tid >> 2, ko4 = (tid & 3) * 8;

    // async-load full 64x768 Q block into LDS
    for (int i = tid; i < 6144; i += 256) {
        int row = i / 96, ko2 = (i % 96) * 8;
        async_b128(Qs + row * 768 + ko2,
                   Q + ((size_t)bb * TB + row) * DD_ + ko2);
    }
    if (tid < 64) { rowmax[tid] = -__builtin_inff(); rowsum[tid] = 0.f; }
    WAIT_ASYNC(0);
    __syncthreads();

    int mrow = (w & 3) * 16, ncol = (w >> 2) * 32;
    v8f o0 = {0.f,0.f,0.f,0.f,0.f,0.f,0.f,0.f};
    v8f o1 = o0;

    for (int kb0 = 0; kb0 < TB; kb0 += 64) {
        // ---- S = Qg @ Kblk^T (double-buffered async K staging) ----
        v8f s0 = {0.f,0.f,0.f,0.f,0.f,0.f,0.f,0.f};
        v8f s1 = s0;
        const unsigned short* gK =
            Kk + ((size_t)bb * TB + kb0 + row4) * DD_ + ko4;
        async_b128(KsB[0] + row4 * 32 + ko4, gK);
        for (int t = 0; t < 24; ++t) {
            int cur = t & 1;
            if (t < 23) {
                async_b128(KsB[cur ^ 1] + row4 * 32 + ko4, gK + (t + 1) * 32);
                WAIT_ASYNC(1);
            } else {
                WAIT_ASYNC(0);
            }
            __syncthreads();
            v16bf a  = load_frag_a(Qs + t * 32, mrow, 768, lane);
            v16bf b0 = load_frag_b(KsB[cur], ncol, 32, lane);
            v16bf b1 = load_frag_b(KsB[cur], ncol + 16, 32, lane);
            s0 = wmma_bf16(a, b0, s0);
            s1 = wmma_bf16(a, b1, s1);
            __syncthreads();
        }
#pragma unroll
        for (int r = 0; r < 8; ++r) {
            S[(mrow + r + 8 * h) * 64 + ncol + lr]      = s0[r] * scale;
            S[(mrow + r + 8 * h) * 64 + ncol + lr + 16] = s1[r] * scale;
        }
        __syncthreads();

        // ---- online softmax update (one thread per row) ----
        if (tid < 64) {
            int row = tid;
            float mo = rowmax[row], mn = mo;
            for (int j = 0; j < 64; ++j) mn = fmaxf(mn, S[row * 64 + j]);
            float corr = __expf(mo - mn);
            float sum = rowsum[row] * corr;
            for (int j = 0; j < 64; ++j) {
                float e = __expf(S[row * 64 + j] - mn);
                Ps[row * 64 + j] = f2bf(e);
                sum += e;
            }
            rowsum[row] = sum;
            rowmax[row] = mn;
            rowscale[row] = corr;
        }
        // ---- stage V tile transposed: Vt[d][k] (64 rows x 8 b128) ----
        for (int i = tid; i < 512; i += 256) {
            int j = i >> 3, dc = (i & 7) * 8;
            unsigned short tmp[8];
            *(uint4*)tmp = *(const uint4*)(V + ((size_t)bb * TB + kb0 + j) * DD_ +
                                           dt * 64 + dc);
#pragma unroll
            for (int e = 0; e < 8; ++e) Vt[(dc + e) * 64 + j] = tmp[e];
        }
        __syncthreads();

        // rescale accumulators, then O += P @ Vblk
#pragma unroll
        for (int r = 0; r < 8; ++r) {
            float c = rowscale[mrow + r + 8 * h];
            o0[r] *= c;
            o1[r] *= c;
        }
#pragma unroll
        for (int kk = 0; kk < 64; kk += 32) {
            v16bf a  = load_frag_a(Ps + kk, mrow, 64, lane);
            v16bf b0 = load_frag_b(Vt + kk, ncol, 64, lane);
            v16bf b1 = load_frag_b(Vt + kk, ncol + 16, 64, lane);
            o0 = wmma_bf16(a, b0, o0);
            o1 = wmma_bf16(a, b1, o1);
        }
        __syncthreads();
    }

    // ---- normalize and overwrite attn rows [0,64) of this batch ----
#pragma unroll
    for (int r = 0; r < 8; ++r) {
        int row = mrow + r + 8 * h;
        float inv = 1.0f / rowsum[row];
        size_t base = ((size_t)bb * TB + row) * DD_ + dt * 64 + ncol + lr;
        attn[base]      = f2bf(o0[r] * inv);
        attn[base + 16] = f2bf(o1[r] * inv);
    }
}

// ---------------------------------------------------------------------------
extern "C" void kernel_launch(void* const* d_in, const int* in_sizes, int n_in,
                              void* d_out, int out_size, void* d_ws, size_t ws_size,
                              hipStream_t stream) {
    (void)in_sizes; (void)n_in; (void)out_size; (void)ws_size;
    const float* x  = (const float*)d_in[0];
    const float* Wq = (const float*)d_in[1];
    const float* bq = (const float*)d_in[2];
    const float* Wk = (const float*)d_in[3];
    const float* bk = (const float*)d_in[4];
    const float* Wv = (const float*)d_in[5];
    const float* bv = (const float*)d_in[6];
    const float* Wo = (const float*)d_in[7];
    const float* bo = (const float*)d_in[8];
    float* out = (float*)d_out;

    const size_t MD = (size_t)MTOT * DD_;
    const size_t WW = (size_t)DD_ * DD_;
    unsigned short* ws  = (unsigned short*)d_ws;
    unsigned short* xb  = ws;
    unsigned short* wqb = xb + MD;
    unsigned short* wkb = wqb + WW;
    unsigned short* wvb = wkb + WW;
    unsigned short* wob = wvb + WW;
    unsigned short* qb  = wob + WW;
    unsigned short* kb  = qb + MD;
    unsigned short* vb  = kb + MD;
    unsigned short* ab  = vb + MD;

    // 1) convert inputs to bf16
    {
        int n4 = (int)(MD / 4);
        cvt_f32_bf16<<<(n4 + 255) / 256, 256, 0, stream>>>(x, xb, n4);
        int w4 = (int)(WW / 4);
        cvt_f32_bf16<<<(w4 + 255) / 256, 256, 0, stream>>>(Wq, wqb, w4);
        cvt_f32_bf16<<<(w4 + 255) / 256, 256, 0, stream>>>(Wk, wkb, w4);
        cvt_f32_bf16<<<(w4 + 255) / 256, 256, 0, stream>>>(Wv, wvb, w4);
        cvt_f32_bf16<<<(w4 + 255) / 256, 256, 0, stream>>>(Wo, wob, w4);
    }

    // 2) Q/K/V projections (bf16 out)
    dim3 gg(MTOT / 64, DD_ / 64);
    gemm_bias<<<gg, 256, 0, stream>>>(xb, wqb, bq, (void*)qb, MTOT, DD_, DD_, 1);
    gemm_bias<<<gg, 256, 0, stream>>>(xb, wkb, bk, (void*)kb, MTOT, DD_, DD_, 1);
    gemm_bias<<<gg, 256, 0, stream>>>(xb, wvb, bv, (void*)vb, MTOT, DD_, DD_, 1);

    // 3) local windowed attention (2 batches * 128 blocks)
    local_attn<<<dim3(2 * NB), 256, 131200, stream>>>(qb, kb, vb, ab);

    // 4) global attention for first 64 queries (overwrites rows [0,64))
    global_attn<<<dim3(2 * 12), 256, 140032, stream>>>(qb, kb, vb, ab);

    // 5) output projection (f32 out)
    gemm_bias<<<gg, 256, 0, stream>>>(ab, wob, bo, (void*)out, MTOT, DD_, DD_, 0);
}
